// TranscriptomePredictor_12867722018959
// MI455X (gfx1250) — matrix operand order repaired and checked by reading, not compiled, not run on hardware
//
#include <hip/hip_runtime.h>
#include <hip/hip_bf16.h>

// ---------------------------------------------------------------------------
// TranscriptomePredictor forward for MI455X (gfx1250, wave32, WMMA).
// All GEMMs (input-proj, mamba in-proj, SSD intra/inter chunk GEMMs,
// out-proj) use v_wmma_f32_16x16x32_f16 with f32 accumulation.
// ---------------------------------------------------------------------------

typedef __attribute__((ext_vector_type(16))) _Float16 v16h;
typedef __attribute__((ext_vector_type(8)))  _Float16 v8h;
typedef __attribute__((ext_vector_type(4)))  _Float16 v4h;
typedef __attribute__((ext_vector_type(8)))  float    v8f;

constexpr int kB     = 4;      // batch
constexpr int kL     = 4096;   // seq len (1 cond + 4095 genes)
constexpr int kNG    = 4095;
constexpr int kDM    = 256;    // d_model
constexpr int kDIN   = 512;    // d_inner
constexpr int kDST   = 128;    // d_state
constexpr int kNH    = 8;      // heads
constexpr int kHD    = 64;     // head dim
constexpr int kCONV  = 768;    // conv dim
constexpr int kDP    = 1288;   // d_in_proj
constexpr int kGFD   = 224;    // gene feature dim
constexpr int kNC    = 64;     // chunks
constexpr int kCH    = 64;     // chunk len
constexpr int kROWS  = kNG + kB; // 4099 rows of the fused feature matrix

__device__ inline float sigm_(float x){ return 1.f/(1.f+__expf(-x)); }
__device__ inline float silu_(float x){ return x*sigm_(x); }
__device__ inline float softplus_(float x){ return (x>20.f)? x : log1pf(__expf(x)); }
__device__ inline float gelu_(float x){ return 0.5f*x*(1.f+erff(x*0.70710678118f)); }

__device__ inline v8f v8zero(){ v8f z = {0.f,0.f,0.f,0.f,0.f,0.f,0.f,0.f}; return z; }

__device__ inline v8h cvt8(float4 a, float4 b){
  v8h h;
  h[0]=(_Float16)a.x; h[1]=(_Float16)a.y; h[2]=(_Float16)a.z; h[3]=(_Float16)a.w;
  h[4]=(_Float16)b.x; h[5]=(_Float16)b.y; h[6]=(_Float16)b.z; h[7]=(_Float16)b.w;
  return h;
}
__device__ inline v4h cvt4(float4 a){
  v4h h;
  h[0]=(_Float16)a.x; h[1]=(_Float16)a.y; h[2]=(_Float16)a.z; h[3]=(_Float16)a.w;
  return h;
}

__device__ inline float block_reduce(float v, float* red){
  int tid = threadIdx.x;
  red[tid] = v; __syncthreads();
  for (int s = blockDim.x>>1; s>0; s>>=1){ if (tid<s) red[tid]+=red[tid+s]; __syncthreads(); }
  float r = red[0]; __syncthreads();
  return r;
}

// ---------------------------------------------------------------------------
// Generic WMMA GEMM:  C[M,N] = A[M,K] * W[N,K]^T (+bias).  A,W,C row-major f32.
// Requires K % 32 == 0.  Block = 256 threads (8 waves); block tile 128x128;
// wave tile 32x64 (2x4 WMMA tiles); K step 32.  f32->f16 staged via float4.
// ---------------------------------------------------------------------------
__global__ __launch_bounds__(256)
void gemm_wmma(const float* __restrict__ A, const float* __restrict__ W,
               const float* __restrict__ bias, float* __restrict__ C,
               int M, int N, int K)
{
  constexpr int BK = 32;
  __shared__ _Float16 As[128][BK+8];   // row stride 80B (16B-aligned)
  __shared__ _Float16 Ws[128][BK+8];

  const int tid  = threadIdx.x;
  const int wave = tid >> 5, lane = tid & 31;
  const int wm   = wave & 3;     // 4 M-strips of 32
  const int wn   = wave >> 2;    // 2 N-strips of 64
  const int bm0  = blockIdx.x * 128;
  const int bn0  = blockIdx.y * 128;

  v8f acc[2][4];
  #pragma unroll
  for (int i=0;i<2;i++)
    #pragma unroll
    for (int j=0;j<4;j++) acc[i][j]=v8zero();

  const int lr  = lane & 15;
  const int kb1 = (lane >> 4) * 16;   // A fragment: K block  0..15 / 16..31
  const int kb2 = (lane >> 4) * 16;   // B fragment: K block  0..15 / 16..31

  // staging coordinates: 2 threads per row, 16 floats each
  const int srow = tid >> 1;
  const int scol = (tid & 1) * 16;

  for (int k0 = 0; k0 < K; k0 += BK) {
    {
      int gr = bm0 + srow;
      float4 f0{},f1{},f2{},f3{};
      if (gr < M) {
        const float4* p = reinterpret_cast<const float4*>(A + (size_t)gr*K + k0 + scol);
        f0 = p[0]; f1 = p[1]; f2 = p[2]; f3 = p[3];
      }
      *reinterpret_cast<v8h*>(&As[srow][scol])     = cvt8(f0,f1);
      *reinterpret_cast<v8h*>(&As[srow][scol + 8]) = cvt8(f2,f3);
    }
    {
      int gr = bn0 + srow;
      float4 f0{},f1{},f2{},f3{};
      if (gr < N) {
        const float4* p = reinterpret_cast<const float4*>(W + (size_t)gr*K + k0 + scol);
        f0 = p[0]; f1 = p[1]; f2 = p[2]; f3 = p[3];
      }
      *reinterpret_cast<v8h*>(&Ws[srow][scol])     = cvt8(f0,f1);
      *reinterpret_cast<v8h*>(&Ws[srow][scol + 8]) = cvt8(f2,f3);
    }
    if (k0 + BK < K) {  // prefetch next tiles (global_prefetch_b8)
      int pr = bm0 + srow;
      if (pr < M) __builtin_prefetch(&A[(size_t)pr*K + k0 + BK + scol], 0, 1);
      int pw = bn0 + srow;
      if (pw < N) __builtin_prefetch(&W[(size_t)pw*K + k0 + BK + scol], 0, 1);
    }
    __syncthreads();

    v16h af[2], bf[4];
    #pragma unroll
    for (int i = 0; i < 2; i++) {
      int row = wm*32 + i*16 + lr;
      #pragma unroll
      for (int j = 0; j < 8; j++) { af[i][j] = As[row][kb1+j]; af[i][8+j] = As[row][8+kb1+j]; }
    }
    #pragma unroll
    for (int i = 0; i < 4; i++) {
      int col = wn*64 + i*16 + lr;
      #pragma unroll
      for (int j = 0; j < 16; j++) bf[i][j] = Ws[col][kb2+j];
    }
    #pragma unroll
    for (int i = 0; i < 2; i++)
      #pragma unroll
      for (int j = 0; j < 4; j++)
        acc[i][j] = __builtin_amdgcn_wmma_f32_16x16x32_f16(
            false, af[i], false, bf[j], (short)0, acc[i][j], false, false);
    __syncthreads();
  }

  const int rofs = (lane >> 4) * 8;
  #pragma unroll
  for (int i = 0; i < 2; i++)
    #pragma unroll
    for (int j = 0; j < 4; j++) {
      int col = bn0 + wn*64 + j*16 + lr;
      #pragma unroll
      for (int r = 0; r < 8; r++) {
        int row = bm0 + wm*32 + i*16 + rofs + r;
        if (row < M && col < N) {
          float v = acc[i][j][r];
          if (bias) v += bias[col];
          C[(size_t)row*N + col] = v;
        }
      }
    }
}

// ---------------------------------------------------------------------------
// Feature building
// ---------------------------------------------------------------------------
__global__ void gene_feat_kernel(const int* __restrict__ chr_idx,
                                 const float* __restrict__ lf,
                                 const float* __restrict__ pf,
                                 const float* __restrict__ gid,
                                 const float* __restrict__ chre,
                                 const float* __restrict__ lw,
                                 const float* __restrict__ lb,
                                 float* __restrict__ Amat)
{
  int g = blockIdx.x, j = threadIdx.x;  // g < 4095, j < 224
  float v;
  if (j < 64)        v = gid[g*64 + j];
  else if (j < 192)  v = pf[g*128 + (j-64)];
  else if (j < 208)  v = chre[chr_idx[g]*16 + (j-192)];
  else {
    int q = j - 208;
    float s = lb[q];
    for (int i = 0; i < 64; i++) s += lf[g*64 + i] * lw[q*64 + i];
    v = gelu_(s);
  }
  Amat[(size_t)g*kGFD + j] = v;
}

__global__ void cond_kernel(const int* __restrict__ pidx,
                            const float* __restrict__ pe,
                            const float* __restrict__ cw,
                            const float* __restrict__ cb,
                            float* __restrict__ Amat)
{
  int b = blockIdx.x, j = threadIdx.x;  // b < 4, j < 224
  const float* e = pe + (size_t)pidx[b]*128;
  float s = cb[j];
  for (int i = 0; i < 128; i++) s += e[i] * cw[j*128 + i];
  Amat[(size_t)(kNG + b)*kGFD + j] = s;
}

// LayerNorm over D=256; writes seq and time-reversed seq.
__global__ __launch_bounds__(256)
void ln_seq(const float* __restrict__ P, const float* __restrict__ g,
            const float* __restrict__ be, float* __restrict__ seq,
            float* __restrict__ seqr)
{
  __shared__ float red[256];
  int row = blockIdx.x;            // b*L + t
  int b = row / kL, t = row % kL;
  int src = (t == 0) ? (kNG + b) : (t - 1);
  int d = threadIdx.x;
  float x = P[(size_t)src*kDM + d];
  float mu = block_reduce(x, red) * (1.f/kDM);
  float c  = x - mu;
  float var = block_reduce(c*c, red) * (1.f/kDM);
  float v = c * __frsqrt_rn(var + 1e-5f) * g[d] + be[d];
  seq[(size_t)row*kDM + d] = v;
  seqr[((size_t)b*kL + (kL-1-t))*kDM + d] = v;
}

// ---------------------------------------------------------------------------
// Mamba elementwise stages
// ---------------------------------------------------------------------------
__global__ void dt_kernel(const float* __restrict__ Z, const float* __restrict__ dtb,
                          float* __restrict__ DT)
{
  int i = blockIdx.x*blockDim.x + threadIdx.x;   // b*L*NH
  if (i >= kB*kL*kNH) return;
  int h = i % kNH; int row = i / kNH;
  float x = Z[(size_t)row*kDP + (kDIN + kCONV) + h] + dtb[h];
  DT[i] = softplus_(x);
}

__global__ void conv_kernel(const float* __restrict__ Z, const float* __restrict__ cw,
                            const float* __restrict__ cb, float* __restrict__ Xs,
                            float* __restrict__ Bm, float* __restrict__ Cm)
{
  int i = blockIdx.x*blockDim.x + threadIdx.x;   // b*L*CONV
  if (i >= kB*kL*kCONV) return;
  int c = i % kCONV; int l = (i / kCONV) % kL; int b = i / (kCONV*kL);
  float acc = cb[c];
  #pragma unroll
  for (int k = 0; k < 4; k++) {
    int t = l - 3 + k;
    if (t >= 0) acc += cw[c*4 + k] * Z[((size_t)b*kL + t)*kDP + kDIN + c];
  }
  float v = silu_(acc);
  size_t row = (size_t)b*kL + l;
  if (c < kDIN)            Xs[row*kDIN + c] = v;
  else if (c < kDIN+kDST)  Bm[row*kDST + (c - kDIN)] = v;
  else                     Cm[row*kDST + (c - kDIN - kDST)] = v;
}

// ---------------------------------------------------------------------------
// SSD intra-chunk: per (b,c,h) block of 128 threads (4 waves).
//   G = C·Bᵀ ; Y_diag = (G∘L)·(X·dt) ; states = (X·dt)ᵀ·(dec·B)
// ---------------------------------------------------------------------------
__global__ __launch_bounds__(128)
void ssd_intra(const float* __restrict__ DT, const float* __restrict__ Xs,
               const float* __restrict__ Bm, const float* __restrict__ Cm,
               const float* __restrict__ A_log, float* __restrict__ Y,
               float* __restrict__ St, float* __restrict__ Cs)
{
  __shared__ float dtl[kCH], acs[kCH], decf[kCH];
  __shared__ _Float16 ChS[kCH][kDST+8];   // row stride 272B (16B-aligned)
  __shared__ _Float16 BhS[kCH][kDST+8];
  __shared__ _Float16 XhS[kCH][kHD+8];    // row stride 144B (16B-aligned)
  __shared__ _Float16 GhS[kCH][kCH+8];

  int bid = blockIdx.x;
  int h = bid % kNH; int c = (bid / kNH) % kNC; int b = bid / (kNH*kNC);
  int t0 = c * kCH;
  int tid = threadIdx.x, wave = tid >> 5, lane = tid & 31;
  const int lr = lane & 15, kb1 = (lane>>4)*16, kb2 = (lane>>4)*16, rofs = (lane>>4)*8;
  float Ah = -__expf(A_log[h]);

  if (tid < kCH) dtl[tid] = DT[((size_t)b*kL + t0 + tid)*kNH + h];
  __syncthreads();
  if (tid < kCH) {
    float s = 0.f;
    for (int i = 0; i <= tid; i++) s += dtl[i] * Ah;
    acs[tid] = s;
  }
  __syncthreads();
  float alast = acs[kCH-1];
  if (tid < kCH) decf[tid] = __expf(alast - acs[tid]);
  // stage C, B (64x128) via float4; 2048 units of 4, 16 units/thread
  for (int u = tid; u < kCH*kDST/4; u += 128) {
    int l = u >> 5, n0 = (u & 31) * 4;
    size_t gb = ((size_t)b*kL + t0 + l)*kDST + n0;
    *reinterpret_cast<v4h*>(&ChS[l][n0]) = cvt4(*reinterpret_cast<const float4*>(Cm + gb));
    *reinterpret_cast<v4h*>(&BhS[l][n0]) = cvt4(*reinterpret_cast<const float4*>(Bm + gb));
  }
  // stage X*dt (64x64); 1024 units
  for (int u = tid; u < kCH*kHD/4; u += 128) {
    int l = u >> 4, p0 = (u & 15) * 4;
    float4 x4 = *reinterpret_cast<const float4*>(Xs + ((size_t)b*kL + t0 + l)*kDIN + h*kHD + p0);
    float d = dtl[l];
    x4.x *= d; x4.y *= d; x4.z *= d; x4.w *= d;
    *reinterpret_cast<v4h*>(&XhS[l][p0]) = cvt4(x4);
  }
  __syncthreads();

  // ---- G = C·Bᵀ  (64x64, K=128), each wave one 16-row strip
  {
    v8f g[4]; for (int i=0;i<4;i++) g[i]=v8zero();
    for (int k0 = 0; k0 < kDST; k0 += 32) {
      v16h af;
      int row = wave*16 + lr;
      #pragma unroll
      for (int j=0;j<8;j++){ af[j]=ChS[row][k0+kb1+j]; af[8+j]=ChS[row][k0+8+kb1+j]; }
      #pragma unroll
      for (int st=0; st<4; st++) {
        v16h bfv; int col = st*16 + lr;
        #pragma unroll
        for (int j=0;j<16;j++) bfv[j]=BhS[col][k0+kb2+j];
        g[st] = __builtin_amdgcn_wmma_f32_16x16x32_f16(false, af, false, bfv, (short)0, g[st], false, false);
      }
    }
    #pragma unroll
    for (int st=0; st<4; st++) {
      int scol = st*16 + lr;
      #pragma unroll
      for (int r=0;r<8;r++) {
        int l = wave*16 + rofs + r;
        float v = (scol <= l) ? g[st][r] * __expf(acs[l]-acs[scol]) : 0.f;
        GhS[l][scol] = (_Float16)v;
      }
    }
  }
  __syncthreads();
  // scale B in place by per-row decay for the states GEMM (vectorized)
  for (int u = tid; u < kCH*kDST/4; u += 128) {
    int l = u >> 5, n0 = (u & 31) * 4;
    v4h v = *reinterpret_cast<v4h*>(&BhS[l][n0]);
    float d = decf[l];
    #pragma unroll
    for (int j=0;j<4;j++) v[j] = (_Float16)((float)v[j] * d);
    *reinterpret_cast<v4h*>(&BhS[l][n0]) = v;
  }
  __syncthreads();

  // ---- Y_diag = G·(X·dt)   (64x64, K=64)
  {
    v8f y[4]; for (int i=0;i<4;i++) y[i]=v8zero();
    for (int k0 = 0; k0 < kCH; k0 += 32) {
      v16h af; int row = wave*16 + lr;
      #pragma unroll
      for (int j=0;j<8;j++){ af[j]=GhS[row][k0+kb1+j]; af[8+j]=GhS[row][k0+8+kb1+j]; }
      #pragma unroll
      for (int pt=0; pt<4; pt++) {
        v16h bfv; int col = pt*16 + lr;
        #pragma unroll
        for (int j=0;j<16;j++) bfv[j]=XhS[k0+kb2+j][col];
        y[pt] = __builtin_amdgcn_wmma_f32_16x16x32_f16(false, af, false, bfv, (short)0, y[pt], false, false);
      }
    }
    #pragma unroll
    for (int pt=0; pt<4; pt++) {
      int p = pt*16 + lr;
      #pragma unroll
      for (int r=0;r<8;r++) {
        int l = wave*16 + rofs + r;
        Y[((size_t)b*kL + t0 + l)*kDIN + h*kHD + p] = y[pt][r];
      }
    }
  }

  // ---- states = (X·dt)ᵀ·(dec·B)   (64x128, K=64)
  {
    v8f s[8]; for (int i=0;i<8;i++) s[i]=v8zero();
    for (int k0 = 0; k0 < kCH; k0 += 32) {
      v16h af; int row = wave*16 + lr;   // row = p
      #pragma unroll
      for (int j=0;j<8;j++){ af[j]=XhS[k0+kb1+j][row]; af[8+j]=XhS[k0+8+kb1+j][row]; }
      #pragma unroll
      for (int nt=0; nt<8; nt++) {
        v16h bfv; int col = nt*16 + lr;
        #pragma unroll
        for (int j=0;j<16;j++) bfv[j]=BhS[k0+kb2+j][col];
        s[nt] = __builtin_amdgcn_wmma_f32_16x16x32_f16(false, af, false, bfv, (short)0, s[nt], false, false);
      }
    }
    size_t base = (((size_t)b*kNC + c)*kNH + h)*(kHD*kDST);
    #pragma unroll
    for (int nt=0; nt<8; nt++) {
      int n = nt*16 + lr;
      #pragma unroll
      for (int r=0;r<8;r++) {
        int p = wave*16 + rofs + r;
        St[base + (size_t)p*kDST + n] = s[nt][r];
      }
    }
  }

  if (tid < kCH) Cs[(((size_t)b*kNC + c)*kNH + h)*kCH + tid] = acs[tid];
}

// Sequential inter-chunk recurrence; St becomes "prev" (state entering chunk).
__global__ __launch_bounds__(256)
void ssd_scan(float* __restrict__ St, const float* __restrict__ Cs)
{
  int bh = blockIdx.x;                 // b*NH + h
  int h = bh % kNH, b = bh / kNH;
  int tid = threadIdx.x;
  float hp[32];
  #pragma unroll
  for (int k=0;k<32;k++) hp[k]=0.f;
  for (int c = 0; c < kNC; c++) {
    float dec = __expf(Cs[(((size_t)b*kNC + c)*kNH + h)*kCH + (kCH-1)]);
    size_t base = (((size_t)b*kNC + c)*kNH + h)*(kHD*kDST);
    #pragma unroll
    for (int k = 0; k < 32; k++) {
      size_t ix = base + tid + k*256;
      float s = St[ix];
      St[ix] = hp[k];
      hp[k] = hp[k]*dec + s;
    }
  }
}

// Y += exp(Acs)·(C·prevᵀ) + D·Xs     per (b,c,h), 128 threads
__global__ __launch_bounds__(128)
void ssd_inter(const float* __restrict__ Cm, const float* __restrict__ St,
               const float* __restrict__ Cs, const float* __restrict__ Xs,
               const float* __restrict__ Dh, float* __restrict__ Y)
{
  __shared__ float acs[kCH];
  __shared__ _Float16 ChS[kCH][kDST+8];
  __shared__ _Float16 PhS[kHD][kDST+8];

  int bid = blockIdx.x;
  int h = bid % kNH; int c = (bid / kNH) % kNC; int b = bid / (kNH*kNC);
  int t0 = c * kCH;
  int tid = threadIdx.x, wave = tid >> 5, lane = tid & 31;
  const int lr = lane & 15, kb1 = (lane>>4)*16, kb2 = (lane>>4)*16, rofs = (lane>>4)*8;

  if (tid < kCH) acs[tid] = Cs[(((size_t)b*kNC + c)*kNH + h)*kCH + tid];
  size_t pbase = (((size_t)b*kNC + c)*kNH + h)*(kHD*kDST);
  for (int u = tid; u < kCH*kDST/4; u += 128) {
    int l = u >> 5, n0 = (u & 31) * 4;
    *reinterpret_cast<v4h*>(&ChS[l][n0]) =
        cvt4(*reinterpret_cast<const float4*>(Cm + ((size_t)b*kL + t0 + l)*kDST + n0));
    *reinterpret_cast<v4h*>(&PhS[l][n0]) =
        cvt4(*reinterpret_cast<const float4*>(St + pbase + (size_t)l*kDST + n0));
  }
  __syncthreads();

  v8f y[4]; for (int i=0;i<4;i++) y[i]=v8zero();
  for (int k0 = 0; k0 < kDST; k0 += 32) {
    v16h af; int row = wave*16 + lr;
    #pragma unroll
    for (int j=0;j<8;j++){ af[j]=ChS[row][k0+kb1+j]; af[8+j]=ChS[row][k0+8+kb1+j]; }
    #pragma unroll
    for (int pt=0; pt<4; pt++) {
      v16h bfv; int col = pt*16 + lr;
      #pragma unroll
      for (int j=0;j<16;j++) bfv[j]=PhS[col][k0+kb2+j];
      y[pt] = __builtin_amdgcn_wmma_f32_16x16x32_f16(false, af, false, bfv, (short)0, y[pt], false, false);
    }
  }
  float dv = Dh[h];
  #pragma unroll
  for (int pt=0; pt<4; pt++) {
    int p = pt*16 + lr;
    #pragma unroll
    for (int r=0;r<8;r++) {
      int l = wave*16 + rofs + r;
      size_t ix = ((size_t)b*kL + t0 + l)*kDIN + h*kHD + p;
      Y[ix] = __expf(acs[l])*y[pt][r] + Y[ix] + dv*Xs[ix];
    }
  }
}

// y = Y * silu(z); RMS-norm over 512 * norm_w
__global__ __launch_bounds__(256)
void gate_norm(const float* __restrict__ Y, const float* __restrict__ Z,
               const float* __restrict__ nw, float* __restrict__ Gy)
{
  __shared__ float red[256];
  int row = blockIdx.x;
  int j0 = threadIdx.x, j1 = threadIdx.x + 256;
  float z0 = Z[(size_t)row*kDP + j0], z1 = Z[(size_t)row*kDP + j1];
  float y0 = Y[(size_t)row*kDIN + j0] * silu_(z0);
  float y1 = Y[(size_t)row*kDIN + j1] * silu_(z1);
  float ss = block_reduce(y0*y0 + y1*y1, red);
  float r = __frsqrt_rn(ss * (1.f/kDIN) + 1e-5f);
  Gy[(size_t)row*kDIN + j0] = y0 * r * nw[j0];
  Gy[(size_t)row*kDIN + j1] = y1 * r * nw[j1];
}

// out[b,g] = dot(yF[b,1+g] + flip(yB)[b,1+g], head_w) + head_b
__global__ __launch_bounds__(256)
void head_combine(const float* __restrict__ yF, const float* __restrict__ yB,
                  const float* __restrict__ hw, const float* __restrict__ hb,
                  float* __restrict__ out)
{
  __shared__ float red[256];
  int i = blockIdx.x;                  // b*NG + g
  int b = i / kNG, g = i % kNG;
  int t = 1 + g;
  int d = threadIdx.x;
  float v = (yF[((size_t)b*kL + t)*kDM + d] +
             yB[((size_t)b*kL + (kL-1-t))*kDM + d]) * hw[d];
  float s = block_reduce(v, red);
  if (d == 0) out[i] = s + hb[0];
}

// ---------------------------------------------------------------------------
// Workspace layout (floats). Total ~86.3M floats (~346 MB).
// ---------------------------------------------------------------------------
constexpr size_t OFF_AMAT = 0;                                       // 4099x224
constexpr size_t OFF_P    = OFF_AMAT + (size_t)4112*kGFD;            // 4099x256
constexpr size_t OFF_SEQ  = OFF_P    + (size_t)4112*kDM;
constexpr size_t OFF_SEQR = OFF_SEQ  + (size_t)kB*kL*kDM;
constexpr size_t OFF_Z    = OFF_SEQR + (size_t)kB*kL*kDM;
constexpr size_t OFF_DT   = OFF_Z    + (size_t)kB*kL*kDP;
constexpr size_t OFF_XS   = OFF_DT   + (size_t)kB*kL*kNH;
constexpr size_t OFF_BM   = OFF_XS   + (size_t)kB*kL*kDIN;
constexpr size_t OFF_CMB  = OFF_BM   + (size_t)kB*kL*kDST;
constexpr size_t OFF_CS   = OFF_CMB  + (size_t)kB*kL*kDST;
constexpr size_t OFF_Y    = OFF_CS   + (size_t)kB*kNC*kNH*kCH;
constexpr size_t OFF_ST   = OFF_Y    + (size_t)kB*kL*kDIN;
constexpr size_t OFF_GY   = OFF_ST   + (size_t)kB*kNC*kNH*kHD*kDST;
constexpr size_t OFF_YF   = OFF_GY   + (size_t)kB*kL*kDIN;
constexpr size_t OFF_YB   = OFF_YF   + (size_t)kB*kL*kDM;

extern "C" void kernel_launch(void* const* d_in, const int* in_sizes, int n_in,
                              void* d_out, int out_size, void* d_ws, size_t ws_size,
                              hipStream_t stream) {
  (void)in_sizes; (void)n_in; (void)out_size; (void)ws_size;
  const int*   pert_idx = (const int*)  d_in[0];
  const int*   chr_idx  = (const int*)  d_in[1];
  const float* locus_f  = (const float*)d_in[2];
  const float* path_f   = (const float*)d_in[3];
  const float* gid_emb  = (const float*)d_in[4];
  const float* chr_emb  = (const float*)d_in[5];
  const float* locus_w  = (const float*)d_in[6];
  const float* locus_b  = (const float*)d_in[7];
  const float* pert_emb = (const float*)d_in[8];
  const float* cond_w   = (const float*)d_in[9];
  const float* cond_b   = (const float*)d_in[10];
  const float* inp_w    = (const float*)d_in[11];
  const float* inp_b    = (const float*)d_in[12];
  const float* ln_g     = (const float*)d_in[13];
  const float* ln_b     = (const float*)d_in[14];
  const float* head_w   = (const float*)d_in[31];
  const float* head_b   = (const float*)d_in[32];

  float* ws   = (float*)d_ws;
  float* AMAT = ws + OFF_AMAT;
  float* P    = ws + OFF_P;
  float* SEQ  = ws + OFF_SEQ;
  float* SEQR = ws + OFF_SEQR;
  float* Z    = ws + OFF_Z;
  float* DT   = ws + OFF_DT;
  float* XS   = ws + OFF_XS;
  float* BM   = ws + OFF_BM;
  float* CMB  = ws + OFF_CMB;
  float* CS   = ws + OFF_CS;
  float* Y    = ws + OFF_Y;
  float* ST   = ws + OFF_ST;
  float* GY   = ws + OFF_GY;
  float* YF   = ws + OFF_YF;
  float* YB   = ws + OFF_YB;

  // --- embedding / input projection (gene rows projected once, shared by batch)
  gene_feat_kernel<<<kNG, kGFD, 0, stream>>>(chr_idx, locus_f, path_f, gid_emb,
                                             chr_emb, locus_w, locus_b, AMAT);
  cond_kernel<<<kB, kGFD, 0, stream>>>(pert_idx, pert_emb, cond_w, cond_b, AMAT);
  gemm_wmma<<<dim3((kROWS+127)/128, (kDM+127)/128), 256, 0, stream>>>(
      AMAT, inp_w, inp_b, P, kROWS, kDM, kGFD);
  ln_seq<<<kB*kL, 256, 0, stream>>>(P, ln_g, ln_b, SEQ, SEQR);

  // --- two mamba directions (identical pipeline, flipped input/weights)
  for (int dir = 0; dir < 2; dir++) {
    int wb = 15 + dir*8;
    const float* in_w    = (const float*)d_in[wb + 0];
    const float* conv_w  = (const float*)d_in[wb + 1];
    const float* conv_b  = (const float*)d_in[wb + 2];
    const float* dt_bias = (const float*)d_in[wb + 3];
    const float* A_log   = (const float*)d_in[wb + 4];
    const float* Dh      = (const float*)d_in[wb + 5];
    const float* norm_w  = (const float*)d_in[wb + 6];
    const float* out_w   = (const float*)d_in[wb + 7];
    const float* x       = dir ? SEQR : SEQ;
    float*       ydir    = dir ? YB   : YF;

    gemm_wmma<<<dim3(kB*kL/128, (kDP+127)/128), 256, 0, stream>>>(
        x, in_w, nullptr, Z, kB*kL, kDP, kDM);
    dt_kernel<<<(kB*kL*kNH+255)/256, 256, 0, stream>>>(Z, dt_bias, DT);
    conv_kernel<<<(kB*kL*kCONV+255)/256, 256, 0, stream>>>(Z, conv_w, conv_b, XS, BM, CMB);
    ssd_intra<<<kB*kNC*kNH, 128, 0, stream>>>(DT, XS, BM, CMB, A_log, Y, ST, CS);
    ssd_scan<<<kB*kNH, 256, 0, stream>>>(ST, CS);
    ssd_inter<<<kB*kNC*kNH, 128, 0, stream>>>(CMB, ST, CS, XS, Dh, Y);
    gate_norm<<<kB*kL, 256, 0, stream>>>(Y, Z, norm_w, GY);
    gemm_wmma<<<dim3(kB*kL/128, (kDM+127)/128), 256, 0, stream>>>(
        GY, out_w, nullptr, ydir, kB*kL, kDM, kDIN);
  }

  head_combine<<<kB*kNG, 256, 0, stream>>>(YF, YB, head_w, head_b, (float*)d_out);
}
